// GINBlock_2491081031684
// MI455X (gfx1250) — matrix operand order repaired and checked by reading, not compiled, not run on hardware
//
#include <hip/hip_runtime.h>

#define D   128
#define NN  100000
#define NE  600000
#define TILES (NN / 16)   // 6250 (exact)
#define WPITCH 136        // bf16 elements per LDS weight row: 272B, 16B-aligned, bank 4n%64

typedef __attribute__((ext_vector_type(16))) __bf16   v16bf;
typedef __attribute__((ext_vector_type(8)))  float    v8f;
typedef __attribute__((ext_vector_type(4)))  unsigned v4u;

// ---- fp32 -> bf16 (round-to-nearest-even) ----
__device__ __forceinline__ unsigned short f2bf_bits(float f) {
  unsigned u = __builtin_bit_cast(unsigned, f);
  u += 0x7FFFu + ((u >> 16) & 1u);
  return (unsigned short)(u >> 16);
}
__device__ __forceinline__ unsigned pack_bf(float lo, float hi) {
#if __has_builtin(__builtin_amdgcn_cvt_pk_bf16_f32)
  typedef __attribute__((ext_vector_type(2))) __bf16 v2bf;
  v2bf p = __builtin_amdgcn_cvt_pk_bf16_f32(lo, hi);
  return __builtin_bit_cast(unsigned, p);
#else
  return (unsigned)f2bf_bits(lo) | ((unsigned)f2bf_bits(hi) << 16);
#endif
}

// ---------------- zero the aggregation buffer ----------------
__global__ __launch_bounds__(256) void zero_kernel(float4* __restrict__ p, int n4) {
  int i = blockIdx.x * blockDim.x + threadIdx.x;
  if (i < n4) p[i] = make_float4(0.f, 0.f, 0.f, 0.f);
}

// ---------------- edge scatter: agg[dst] += x[src] ----------------
// 32 lanes per edge, 4 floats per lane. x and agg both fit in the 192MB L2,
// so the random gather + fp32 atomics stay on-chip.
__global__ __launch_bounds__(256) void scatter_add_kernel(
    const float* __restrict__ x,
    const long long* __restrict__ src,
    const long long* __restrict__ dst,
    float* __restrict__ agg) {
  int g = blockIdx.x * blockDim.x + threadIdx.x;
  int e = g >> 5;
  if (e >= NE) return;
  int j = (g & 31) << 2;
  long long s = src[e];
  long long d = dst[e];
  const float4 v = *(const float4*)(x + (size_t)s * D + j);
  float* p = agg + (size_t)d * D + j;
  __hip_atomic_fetch_add(p + 0, v.x, __ATOMIC_RELAXED, __HIP_MEMORY_SCOPE_AGENT);
  __hip_atomic_fetch_add(p + 1, v.y, __ATOMIC_RELAXED, __HIP_MEMORY_SCOPE_AGENT);
  __hip_atomic_fetch_add(p + 2, v.z, __ATOMIC_RELAXED, __HIP_MEMORY_SCOPE_AGENT);
  __hip_atomic_fetch_add(p + 3, v.w, __ATOMIC_RELAXED, __HIP_MEMORY_SCOPE_AGENT);
}

// ---------------- fused 2-layer MLP on (x + agg) per node ----------------
// out = (RELU_OUT? relu : id)( relu((x+agg) @ W1 + b1) @ W2 + b2 )
// One wave handles 16 nodes. bf16 WMMA, fp32 accumulate.
template <bool RELU_OUT>
__global__ __launch_bounds__(256) void fused_gin_mlp(
    const float* __restrict__ xin, const float* __restrict__ agg,
    const float* __restrict__ W1, const float* __restrict__ b1,
    const float* __restrict__ W2, const float* __restrict__ b2,
    float* __restrict__ out) {
  // Weights transposed in LDS: sW[n*WPITCH + k]. Rows are 16B-aligned so
  // B-fragments load as two ds_load_b128; lane n hits banks 4n..4n+3 mod 64.
  __shared__ __bf16 sW1[D * WPITCH];
  __shared__ __bf16 sW2[D * WPITCH];
  __shared__ float  sb1[D];
  __shared__ float  sb2[D];
  __shared__ __bf16 sH[8][16 * D];   // per-wave layer-1 output tile (16x128 bf16)

  // cooperative transposed weight staging, packed fp32->bf16 pairs
  for (int p = threadIdx.x; p < D * D / 2; p += 256) {
    int n = p & 127;
    int k = (p >> 7) << 1;
    *(unsigned*)&sW1[n * WPITCH + k] = pack_bf(W1[k * D + n], W1[(k + 1) * D + n]);
    *(unsigned*)&sW2[n * WPITCH + k] = pack_bf(W2[k * D + n], W2[(k + 1) * D + n]);
  }
  if (threadIdx.x < D) {
    sb1[threadIdx.x] = b1[threadIdx.x];
    sb2[threadIdx.x] = b2[threadIdx.x];
  }
  __syncthreads();

  const int wave = threadIdx.x >> 5;
  const int lane = threadIdx.x & 31;
  const int m    = lane & 15;   // row within tile (A) / col within tile (B,C)
  const int h    = lane >> 4;   // half-wave K split
  const int tile = blockIdx.x * 8 + wave;
  if (tile >= TILES) return;
  const int node0 = tile << 4;
  const size_t rowbase = (size_t)(node0 + m) * D;

  union AB { unsigned u[8]; v4u q[2]; v16bf v; };

  // ---- A fragments, layer 1: z = x + agg (ISA 16-bit A layout, float4 loads)
  AB a1[4];
#pragma unroll
  for (int kk = 0; kk < 4; ++kk) {
#pragma unroll
    for (int hf = 0; hf < 2; ++hf) {
      const int K = kk * 32 + 16 * hf + 8 * h;       // 8-element contiguous run
      float4 x0 = *(const float4*)(xin + rowbase + K);
      float4 x1 = *(const float4*)(xin + rowbase + K + 4);
      float4 g0 = *(const float4*)(agg + rowbase + K);
      float4 g1 = *(const float4*)(agg + rowbase + K + 4);
      a1[kk].u[hf * 4 + 0] = pack_bf(x0.x + g0.x, x0.y + g0.y);
      a1[kk].u[hf * 4 + 1] = pack_bf(x0.z + g0.z, x0.w + g0.w);
      a1[kk].u[hf * 4 + 2] = pack_bf(x1.x + g1.x, x1.y + g1.y);
      a1[kk].u[hf * 4 + 3] = pack_bf(x1.z + g1.z, x1.w + g1.w);
    }
  }

  __bf16* myH = &sH[wave][0];

  // ---- layer 1: h1 = relu(z @ W1 + b1) -> LDS (bf16)
#pragma unroll 1
  for (int j = 0; j < 8; ++j) {
    const int n = j * 16 + m;
    const __bf16* wrow = &sW1[n * WPITCH];
    v8f acc = {};
#pragma unroll
    for (int kk = 0; kk < 4; ++kk) {
      AB b;
      const v4u* wq = (const v4u*)(wrow + kk * 32 + 8 * h);
      b.q[0] = wq[0];
      b.q[1] = wq[2];
      acc = __builtin_amdgcn_wmma_f32_16x16x32_bf16(
          false, a1[kk].v, false, b.v, (short)0, acc, false, false);
    }
    const float bias = sb1[n];
#pragma unroll
    for (int r = 0; r < 8; ++r) {
      float v = fmaxf(acc[r] + bias, 0.f);
      myH[(r + 8 * h) * D + n] = __builtin_bit_cast(__bf16, f2bf_bits(v));
    }
  }

  // ---- A fragments, layer 2 from LDS (in-order per wave; b128 runs)
  AB a2[4];
#pragma unroll
  for (int kk = 0; kk < 4; ++kk) {
#pragma unroll
    for (int hf = 0; hf < 2; ++hf) {
      const int K = kk * 32 + 16 * hf + 8 * h;
      a2[kk].q[hf] = *(const v4u*)(myH + m * D + K);
    }
  }

  // ---- layer 2: out = h1 @ W2 + b2 (+ optional outer relu)
#pragma unroll 1
  for (int j = 0; j < 8; ++j) {
    const int n = j * 16 + m;
    const __bf16* wrow = &sW2[n * WPITCH];
    v8f acc = {};
#pragma unroll
    for (int kk = 0; kk < 4; ++kk) {
      AB b;
      const v4u* wq = (const v4u*)(wrow + kk * 32 + 8 * h);
      b.q[0] = wq[0];
      b.q[1] = wq[2];
      acc = __builtin_amdgcn_wmma_f32_16x16x32_bf16(
          false, a2[kk].v, false, b.v, (short)0, acc, false, false);
    }
    const float bias = sb2[n];
#pragma unroll
    for (int r = 0; r < 8; ++r) {
      float v = acc[r] + bias;
      if (RELU_OUT) v = fmaxf(v, 0.f);
      out[(size_t)(node0 + r + 8 * h) * D + n] = v;  // coalesced 16-float rows
    }
  }
}

extern "C" void kernel_launch(void* const* d_in, const int* in_sizes, int n_in,
                              void* d_out, int out_size, void* d_ws, size_t ws_size,
                              hipStream_t stream) {
  const float*     x   = (const float*)d_in[0];
  const long long* ei  = (const long long*)d_in[1];   // int64 edge_index [2][NE]
  const float*     W1a = (const float*)d_in[2];
  const float*     b1a = (const float*)d_in[3];
  const float*     W1b = (const float*)d_in[4];
  const float*     b1b = (const float*)d_in[5];
  const float*     W2a = (const float*)d_in[6];
  const float*     b2a = (const float*)d_in[7];
  const float*     W2b = (const float*)d_in[8];
  const float*     b2b = (const float*)d_in[9];
  float* out  = (float*)d_out;
  float* agg  = (float*)d_ws;                 // 100000*128 fp32 = 51.2 MB
  float* hbuf = agg + (size_t)NN * D;         // 51.2 MB more
  const long long* src = ei;
  const long long* dst = ei + NE;

  const int n4 = NN * D / 4;
  dim3 blk(256);

  // Layer 1
  zero_kernel<<<(n4 + 255) / 256, blk, 0, stream>>>((float4*)agg, n4);
  scatter_add_kernel<<<(NE * 32 + 255) / 256, blk, 0, stream>>>(x, src, dst, agg);
  fused_gin_mlp<true><<<(TILES + 7) / 8, blk, 0, stream>>>(x, agg, W1a, b1a, W1b, b1b, hbuf);

  // Layer 2
  zero_kernel<<<(n4 + 255) / 256, blk, 0, stream>>>((float4*)agg, n4);
  scatter_add_kernel<<<(NE * 32 + 255) / 256, blk, 0, stream>>>(hbuf, src, dst, agg);
  fused_gin_mlp<false><<<(TILES + 7) / 8, blk, 0, stream>>>(hbuf, agg, W2a, b2a, W2b, b2b, out);
}
